// TransformerModel_43138651521033
// MI455X (gfx1250) — compile-verified, matrix operand order checked by reading, not run on hardware
//
#include <hip/hip_runtime.h>
#include <hip/hip_bf16.h>

// ---------------------------------------------------------------------------
// CDNA5 / gfx1250 transformer forward.  All GEMMs and attention contractions
// go through v_wmma_f32_16x16x32_f16 (wave32).  Activations live in f16 in
// workspace; accumulation and norms are f32.  GEMM mainloop is double-
// buffered (global->regs -> LDS ping-pong) with global_prefetch for the
// tile two steps ahead.
// ---------------------------------------------------------------------------

typedef __attribute__((ext_vector_type(16))) _Float16 v16h;
typedef __attribute__((ext_vector_type(8)))  float    v8f;

constexpr int Bz  = 256;   // batch
constexpr int Ss  = 128;   // src seq
constexpr int TQq = 64;    // tgt seq
constexpr int Dd  = 256;   // model dim
constexpr int Hh  = 8;     // heads
constexpr int HDd = 32;    // head dim
constexpr int FFf = 1024;  // ffn dim
constexpr int NE  = Ss * Bz;   // 32768 encoder tokens
constexpr int ND  = TQq * Bz;  // 16384 decoder tokens

// ---------------- WMMA helpers --------------------------------------------

__device__ __forceinline__ v8f wmma_f16(v16h a, v16h b, v8f c) {
  // D = A(16x32 f16) * B(32x16 f16) + C(16x16 f32)
  return __builtin_amdgcn_wmma_f32_16x16x32_f16(
      /*neg_a*/false, a, /*neg_b*/false, b,
      /*c_mod*/(short)0, c, /*reuse_a*/false, /*reuse_b*/false);
}

// 16-bit A-matrix 16x32 layout (ISA 7.12.2): lane<16 -> M=lane, K in
// {0..7,16..23}; lane>=16 -> M=lane-16, K in {8..15,24..31}.
// Used for A operands, and for B operands whose source memory is [N][K]
// row-major (i.e. W stored as (out,in): A @ W^T).
__device__ __forceinline__ v16h frag_row(const _Float16* t, int ld, int lane) {
  int m  = lane & 15;
  int kb = (lane < 16) ? 0 : 8;
  const _Float16* p = t + (size_t)m * ld;
  v16h f;
#pragma unroll
  for (int e = 0; e < 8; ++e) { f[e] = p[kb + e]; f[8 + e] = p[kb + 16 + e]; }
  return f;
}

// B operand whose source memory is [K][N] row-major (e.g. V in attention):
// lane holds column n with the K pattern -> strided reads.
__device__ __forceinline__ v16h frag_col(const _Float16* t, int ld, int lane) {
  int n  = lane & 15;
  int kb = (lane < 16) ? 0 : 8;
  v16h f;
#pragma unroll
  for (int e = 0; e < 8; ++e) {
    f[e]     = t[(size_t)(kb + e) * ld + n];
    f[8 + e] = t[(size_t)(kb + 16 + e) * ld + n];
  }
  return f;
}

// ---------------- generic GEMM: C[M,N] = A[M,K] @ W[N,K]^T + bias ----------
// block: 256 threads = 8 waves; block tile 128x64, wave tile 32x32.
// Double-buffered: tile k+1 is loaded to registers while tile k feeds WMMA,
// then stored to the other LDS buffer; one barrier per K-step.

__global__ __launch_bounds__(256) void k_gemm(
    const _Float16* __restrict__ A, const _Float16* __restrict__ W,
    const float* __restrict__ bias, _Float16* __restrict__ C,
    int M, int N, int K, int relu) {
  __shared__ _Float16 As[2][128 * 32];
  __shared__ _Float16 Ws[2][64 * 32];
  const int tid = threadIdx.x;
  const int lane = tid & 31, wave = tid >> 5;
  const int wm = wave >> 1, wn = wave & 1;
  const int bm = blockIdx.x * 128, bn = blockIdx.y * 64;

  // staging addresses for this thread (16 halfs of A, 8 halfs of W per step)
  const int ar = tid >> 1, aseg = (tid & 1) * 16;
  const int wr = tid >> 2, wseg = (tid & 3) * 8;
  const _Float16* Abase = A + (size_t)(bm + ar) * K + aseg;
  const _Float16* Wbase = W + (size_t)(bn + wr) * K + wseg;

  uint4 ra0, ra1, rw0;
  auto gload = [&](int k0) {
    const uint4* ap = (const uint4*)(Abase + k0);
    ra0 = ap[0];
    ra1 = ap[1];
    rw0 = *(const uint4*)(Wbase + k0);
  };
  auto sstore = [&](int buf) {
    uint4* ad = (uint4*)(As[buf] + ar * 32 + aseg);
    ad[0] = ra0;
    ad[1] = ra1;
    *(uint4*)(Ws[buf] + wr * 32 + wseg) = rw0;
  };

  v8f acc[2][2];
#pragma unroll
  for (int i = 0; i < 2; ++i)
#pragma unroll
    for (int j = 0; j < 2; ++j) acc[i][j] = (v8f){0,0,0,0,0,0,0,0};

  const int nk = K >> 5;
  gload(0);
  sstore(0);
  __syncthreads();

  for (int it = 0; it < nk; ++it) {
    // issue next tile's global loads before touching LDS
    if (it + 1 < nk) gload((it + 1) << 5);
    // pull the tile after that toward the caches (gfx1250 global_prefetch)
    if (it + 2 < nk) {
      __builtin_prefetch(Abase + ((it + 2) << 5), 0, 3);
      __builtin_prefetch(Wbase + ((it + 2) << 5), 0, 3);
    }

    const int buf = it & 1;
    v16h a0 = frag_row(As[buf] + (wm * 32 + 0)  * 32, 32, lane);
    v16h a1 = frag_row(As[buf] + (wm * 32 + 16) * 32, 32, lane);
    v16h b0 = frag_row(Ws[buf] + (wn * 32 + 0)  * 32, 32, lane);
    v16h b1 = frag_row(Ws[buf] + (wn * 32 + 16) * 32, 32, lane);
    acc[0][0] = wmma_f16(a0, b0, acc[0][0]);
    acc[0][1] = wmma_f16(a0, b1, acc[0][1]);
    acc[1][0] = wmma_f16(a1, b0, acc[1][0]);
    acc[1][1] = wmma_f16(a1, b1, acc[1][1]);

    if (it + 1 < nk) sstore(buf ^ 1);
    __syncthreads();
  }

  const int ln = lane & 15, hi = lane >> 4;
#pragma unroll
  for (int mi = 0; mi < 2; ++mi)
#pragma unroll
    for (int ni = 0; ni < 2; ++ni) {
      int col = bn + wn * 32 + ni * 16 + ln;
      float bv = bias[col];
#pragma unroll
      for (int j = 0; j < 8; ++j) {
        int row = bm + wm * 32 + mi * 16 + j + hi * 8;
        float v = acc[mi][ni][j] + bv;
        if (relu) v = fmaxf(v, 0.f);
        C[(size_t)row * N + col] = (_Float16)v;
      }
    }
}

// ---------------- fused attention per (b,h) --------------------------------
// scores = Q K^T * scale (+causal), softmax in registers, O = P V via WMMA.
// Token-major rows: token (s,b) -> row s*Bz + b.

__global__ __launch_bounds__(256) void k_attn(
    const _Float16* __restrict__ Q, int qs, int qo,
    const _Float16* __restrict__ Kp, int ks, int ko,
    const _Float16* __restrict__ Vp, int vs, int vo,
    _Float16* __restrict__ O, int os,
    int SQ, int SK, int causal) {
  __shared__ _Float16 Ks[128 * 32];
  __shared__ _Float16 Vs[128 * 32];
  __shared__ _Float16 Ps[8 * 16 * 128];  // per-wave P strip (16 x SKmax)

  const int b = blockIdx.x / Hh, h = blockIdx.x % Hh;
  const int tid = threadIdx.x, lane = tid & 31, wave = tid >> 5;
  const int ln = lane & 15, hi = lane >> 4;

  // stage K and V head slices into LDS: [SK][32]
  for (int i = tid; i < SK * 32; i += 256) {
    int t = i >> 5, d = i & 31;
    size_t kr = ((size_t)t * Bz + b) * ks + ko + h * HDd + d;
    size_t vr = ((size_t)t * Bz + b) * vs + vo + h * HDd + d;
    Ks[i] = Kp[kr];
    Vs[i] = Vp[vr];
  }
  __syncthreads();

  const int sq0 = wave * 16;
  const bool active = (sq0 < SQ);   // wave-uniform -> EXEC all ones inside
  const int nkt = SK >> 4;

  v8f acc[8];
#pragma unroll
  for (int kt = 0; kt < 8; ++kt) acc[kt] = (v8f){0,0,0,0,0,0,0,0};

  if (active) {
    // Q fragment straight from global (A layout)
    const int kb = hi ? 8 : 0;
    const _Float16* qrow =
        Q + ((size_t)(sq0 + ln) * Bz + b) * qs + qo + h * HDd;
    v16h qa;
#pragma unroll
    for (int e = 0; e < 8; ++e) { qa[e] = qrow[kb + e]; qa[8 + e] = qrow[kb + 16 + e]; }

    // scores: B operand = K^T, i.e. rows of K -> frag_row on Ks
#pragma unroll
    for (int kt = 0; kt < 8; ++kt)
      if (kt < nkt) {
        v16h kf = frag_row(Ks + (size_t)(kt * 16) * 32, 32, lane);
        acc[kt] = wmma_f16(qa, kf, acc[kt]);
      }

    const float scale = 0.17677669529663687f;  // 1/sqrt(32)
#pragma unroll
    for (int j = 0; j < 8; ++j) {
      int qidx = sq0 + j + hi * 8;
      float rmax = -1e30f;
#pragma unroll
      for (int kt = 0; kt < 8; ++kt)
        if (kt < nkt) {
          float s = acc[kt][j] * scale;
          int key = kt * 16 + ln;
          if (causal && key > qidx) s = -1e9f;  // matches reference mask
          acc[kt][j] = s;
          rmax = fmaxf(rmax, s);
        }
      // reduce within 16-lane half (rows j / j+8 live in disjoint halves)
#pragma unroll
      for (int m = 1; m < 16; m <<= 1)
        rmax = fmaxf(rmax, __shfl_xor(rmax, m, 32));
      float rsum = 0.f;
#pragma unroll
      for (int kt = 0; kt < 8; ++kt)
        if (kt < nkt) {
          float p = __expf(acc[kt][j] - rmax);
          acc[kt][j] = p;
          rsum += p;
        }
#pragma unroll
      for (int m = 1; m < 16; m <<= 1) rsum += __shfl_xor(rsum, m, 32);
      float inv = 1.f / rsum;
#pragma unroll
      for (int kt = 0; kt < 8; ++kt)
        if (kt < nkt)
          Ps[wave * 2048 + (j + hi * 8) * 128 + kt * 16 + ln] =
              (_Float16)(acc[kt][j] * inv);
    }
  }
  __syncthreads();

  if (active) {
    v8f o0 = (v8f){0,0,0,0,0,0,0,0}, o1 = (v8f){0,0,0,0,0,0,0,0};
    const _Float16* Pw = Ps + wave * 2048;
    for (int k0 = 0; k0 < SK; k0 += 32) {
      v16h pa  = frag_row(Pw + k0, 128, lane);
      v16h vb0 = frag_col(Vs + (size_t)k0 * 32 + 0,  32, lane);
      v16h vb1 = frag_col(Vs + (size_t)k0 * 32 + 16, 32, lane);
      o0 = wmma_f16(pa, vb0, o0);
      o1 = wmma_f16(pa, vb1, o1);
    }
#pragma unroll
    for (int j = 0; j < 8; ++j) {
      int s = sq0 + j + hi * 8;
      size_t base = ((size_t)s * Bz + b) * os + h * HDd;
      O[base + ln]      = (_Float16)o0[j];
      O[base + 16 + ln] = (_Float16)o1[j];
    }
  }
}

// ---------------- residual add + LayerNorm (wave per row, D=256) -----------

__global__ __launch_bounds__(256) void k_add_ln(
    const _Float16* __restrict__ A, const _Float16* __restrict__ G,
    const float* __restrict__ w, const float* __restrict__ bb,
    _Float16* __restrict__ Out) {
  const int wave = threadIdx.x >> 5, lane = threadIdx.x & 31;
  const int row = blockIdx.x * 8 + wave;
  const size_t base = (size_t)row * Dd;
  float x[8], sum = 0.f;
#pragma unroll
  for (int i = 0; i < 8; ++i) {
    int c = i * 32 + lane;
    x[i] = (float)A[base + c] + (float)G[base + c];
    sum += x[i];
  }
#pragma unroll
  for (int m = 1; m < 32; m <<= 1) sum += __shfl_xor(sum, m, 32);
  float mu = sum * (1.f / Dd);
  float vs = 0.f;
#pragma unroll
  for (int i = 0; i < 8; ++i) { float d = x[i] - mu; vs += d * d; }
#pragma unroll
  for (int m = 1; m < 32; m <<= 1) vs += __shfl_xor(vs, m, 32);
  float rstd = rsqrtf(vs * (1.f / Dd) + 1e-5f);
#pragma unroll
  for (int i = 0; i < 8; ++i) {
    int c = i * 32 + lane;
    Out[base + c] = (_Float16)((x[i] - mu) * rstd * w[c] + bb[c]);
  }
}

// ---------------- embeddings ----------------------------------------------

__global__ void k_enc_embed(const float* __restrict__ src,
                            const float* __restrict__ w,
                            const float* __restrict__ bb,
                            _Float16* __restrict__ X) {
  int idx = blockIdx.x * 256 + threadIdx.x;  // NE * Dd total
  int d = idx & (Dd - 1);
  int tok = idx >> 8;
  int s = tok / Bz, b = tok % Bz;  // tok = s*Bz + b
  float val = src[((size_t)b * Ss + s) * 4 + 0];
  int pair = d >> 1;
  float div = __expf((float)(2 * pair) * (-9.210340371976184f / (float)Dd));
  float ang = (float)s * div;
  float pe = (d & 1) ? __cosf(ang) : __sinf(ang);
  X[idx] = (_Float16)(val * w[d] + bb[d] + pe);
}

__global__ void k_dec_embed(const float* __restrict__ dec,
                            const float* __restrict__ w,
                            const float* __restrict__ bb,
                            _Float16* __restrict__ X) {
  int idx = blockIdx.x * 256 + threadIdx.x;  // ND * Dd total
  int d = idx & (Dd - 1);
  int tok = idx >> 8;
  int t = tok / Bz, b = tok % Bz;
  float val = dec[(size_t)b * TQq + t];
  X[idx] = (_Float16)(val * w[d] + bb[d]);
}

// ---------------- output projection (N=1): wave-per-row dot ----------------

__global__ __launch_bounds__(256) void k_outproj(
    const _Float16* __restrict__ X, const float* __restrict__ w,
    const float* __restrict__ bb, float* __restrict__ out) {
  const int wave = threadIdx.x >> 5, lane = threadIdx.x & 31;
  const int row = blockIdx.x * 8 + wave;
  const size_t base = (size_t)row * Dd;
  float s = 0.f;
#pragma unroll
  for (int i = 0; i < 8; ++i) {
    int c = i * 32 + lane;
    s += (float)X[base + c] * w[c];
  }
#pragma unroll
  for (int m = 1; m < 32; m <<= 1) s += __shfl_xor(s, m, 32);
  if (lane == 0) out[row] = s + bb[0];
}

// ---------------- hist + IDM scan + fusion (one block, lane = batch) -------

__global__ __launch_bounds__(256) void k_tail(
    const float* __restrict__ src, const float* __restrict__ his,
    const float* __restrict__ out_tok, const float* __restrict__ fw,
    const float* __restrict__ fb, float* __restrict__ out) {
  const int b = threadIdx.x;
  __shared__ float vlast;
  const float S0c = 2.f, TGc = 1.5f, AAc = 1.f, BBc = 1.5f, VDc = 30.f,
              DTc = 0.1f;
  const float two_sab = 2.f * sqrtf(AAc * BBc);

  const float* lr = src + ((size_t)b * Ss + (Ss - 1)) * 4;
  float y = lr[0], v = lr[1], sgap = lr[2], dvv = lr[3];
  float sx = S0c + fmaxf(0.f, v * TGc + v * dvv / two_sab);
  float vr = v / VDc, v4 = vr * vr * vr * vr;
  float r = sx / sgap;
  float af = AAc * (1.f - v4 - r * r);
  float v0 = fmaxf(v + af * DTc, 0.f);
  if (b == Bz - 1) vlast = v0;  // reference quirk: y0 uses v0[-1] for all b
  __syncthreads();
  float y0 = y + vlast * DTc;

  float ylast = lr[0];
  float dv = ylast - src[((size_t)b * Ss + (Ss - 2)) * 4 + 0];
  float w0 = fw[0], w1 = fw[1], w2 = fw[2], fbv = fb[0];

  float vj = v0, yj = y0, idm = y0;
  for (int t = 0; t < TQq; ++t) {
    if (t > 0) {
      const float* hj = his + ((size_t)b * TQq + (t - 1)) * 2;
      float dvj = hj[1] - vj;
      float sj = hj[0] - yj;
      float sxj = S0c + fmaxf(0.f, vj * TGc + vj * dvj / two_sab);
      float vrj = vj / VDc, v4j = vrj * vrj * vrj * vrj;
      float rr = sxj / sj;
      float accj = AAc * (1.f - v4j - rr * rr);
      float v2 = vj + accj * DTc;
      v2 = (v2 <= 0.f) ? 0.f : v2;
      yj += v2 * DTc;
      vj = v2;
      idm = yj;
    }
    float hist = ylast + dv * (float)(t + 1);
    float o = out_tok[(size_t)t * Bz + b];
    out[(size_t)b * TQq + t] = w0 * o + w1 * hist + w2 * idm + fbv;
  }
}

// ---------------- f32 -> f16 weight conversion -----------------------------

__global__ void k_cvt(const float* __restrict__ s, _Float16* __restrict__ d,
                      int n) {
  int i = blockIdx.x * 256 + threadIdx.x;
  if (i < n) d[i] = (_Float16)s[i];
}

// ---------------------------------------------------------------------------
// Host orchestration.
// Input flattening assumed depth-first in setup_inputs() insertion order:
//  0 src_inputs  1 dec_input  2 target_mask  3 his_labels
//  4 enc_emb_w 5 enc_emb_b 6 dec_emb_w 7 dec_emb_b 8 out_w 9 out_b
// 10 fus_w 11 fus_b
// enc layer l (base 12 + 12*l): in_w,in_b,out_w,out_b,l1_w,l1_b,l2_w,l2_b,
//                               ln1_w,ln1_b,ln2_w,ln2_b
// dec layer l (base 36 + 18*l): sa_in_w,sa_in_b,sa_out_w,sa_out_b,
//   ca_in_w,ca_in_b,ca_out_w,ca_out_b,l1_w,l1_b,l2_w,l2_b,
//   ln1_w,ln1_b,ln2_w,ln2_b,ln3_w,ln3_b
// ---------------------------------------------------------------------------

extern "C" void kernel_launch(void* const* d_in, const int* in_sizes, int n_in,
                              void* d_out, int out_size, void* d_ws,
                              size_t ws_size, hipStream_t stream) {
  (void)in_sizes; (void)n_in; (void)out_size; (void)ws_size;
  char* ws = (char*)d_ws;
  size_t off = 0;
  auto alloc = [&](size_t bytes) -> void* {
    void* p = ws + off;
    off = (off + bytes + 255) & ~(size_t)255;
    return p;
  };
  auto F32 = [&](int i) { return (const float*)d_in[i]; };

  // activation buffers (f16)
  _Float16* xe     = (_Float16*)alloc((size_t)NE * Dd * 2);
  _Float16* xd     = (_Float16*)alloc((size_t)ND * Dd * 2);
  _Float16* bigA   = (_Float16*)alloc((size_t)NE * FFf * 2);  // qkv / ffn / kv
  _Float16* attn_o = (_Float16*)alloc((size_t)NE * Dd * 2);
  _Float16* gout   = (_Float16*)alloc((size_t)NE * Dd * 2);
  _Float16* qd     = (_Float16*)alloc((size_t)ND * Dd * 2);
  float*    out_tok = (float*)alloc((size_t)ND * 4);

  auto cvt = [&](int idx, size_t n) -> _Float16* {
    _Float16* p = (_Float16*)alloc(n * 2);
    k_cvt<<<(int)((n + 255) / 256), 256, 0, stream>>>(F32(idx), p, (int)n);
    return p;
  };

  // weight conversion (f32 -> f16) into workspace
  _Float16 *e_in[2], *e_out[2], *e_l1[2], *e_l2[2];
  for (int l = 0; l < 2; ++l) {
    int base = 12 + 12 * l;
    e_in[l]  = cvt(base + 0, (size_t)3 * Dd * Dd);
    e_out[l] = cvt(base + 2, (size_t)Dd * Dd);
    e_l1[l]  = cvt(base + 4, (size_t)FFf * Dd);
    e_l2[l]  = cvt(base + 6, (size_t)Dd * FFf);
  }
  _Float16 *d_sa_in[2], *d_sa_out[2], *d_ca_in[2], *d_ca_out[2], *d_l1[2],
      *d_l2[2];
  for (int l = 0; l < 2; ++l) {
    int base = 36 + 18 * l;
    d_sa_in[l]  = cvt(base + 0, (size_t)3 * Dd * Dd);
    d_sa_out[l] = cvt(base + 2, (size_t)Dd * Dd);
    d_ca_in[l]  = cvt(base + 4, (size_t)3 * Dd * Dd);
    d_ca_out[l] = cvt(base + 6, (size_t)Dd * Dd);
    d_l1[l]     = cvt(base + 8, (size_t)FFf * Dd);
    d_l2[l]     = cvt(base + 10, (size_t)Dd * FFf);
  }

  // embeddings
  k_enc_embed<<<NE * Dd / 256, 256, 0, stream>>>(F32(0), F32(4), F32(5), xe);
  k_dec_embed<<<ND * Dd / 256, 256, 0, stream>>>(F32(1), F32(6), F32(7), xd);

  // ---- encoder ----
  for (int l = 0; l < 2; ++l) {
    int base = 12 + 12 * l;
    k_gemm<<<dim3(NE / 128, (3 * Dd) / 64), 256, 0, stream>>>(
        xe, e_in[l], F32(base + 1), bigA, NE, 3 * Dd, Dd, 0);
    k_attn<<<Bz * Hh, 256, 0, stream>>>(bigA, 3 * Dd, 0, bigA, 3 * Dd, Dd,
                                        bigA, 3 * Dd, 2 * Dd, attn_o, Dd,
                                        Ss, Ss, 0);
    k_gemm<<<dim3(NE / 128, Dd / 64), 256, 0, stream>>>(
        attn_o, e_out[l], F32(base + 3), gout, NE, Dd, Dd, 0);
    k_add_ln<<<NE / 8, 256, 0, stream>>>(xe, gout, F32(base + 8),
                                         F32(base + 9), xe);
    k_gemm<<<dim3(NE / 128, FFf / 64), 256, 0, stream>>>(
        xe, e_l1[l], F32(base + 5), bigA, NE, FFf, Dd, 1);
    k_gemm<<<dim3(NE / 128, Dd / 64), 256, 0, stream>>>(
        bigA, e_l2[l], F32(base + 7), gout, NE, Dd, FFf, 0);
    k_add_ln<<<NE / 8, 256, 0, stream>>>(xe, gout, F32(base + 10),
                                         F32(base + 11), xe);
  }

  // ---- decoder ----
  for (int l = 0; l < 2; ++l) {
    int base = 36 + 18 * l;
    // masked self-attention
    k_gemm<<<dim3(ND / 128, (3 * Dd) / 64), 256, 0, stream>>>(
        xd, d_sa_in[l], F32(base + 1), bigA, ND, 3 * Dd, Dd, 0);
    k_attn<<<Bz * Hh, 256, 0, stream>>>(bigA, 3 * Dd, 0, bigA, 3 * Dd, Dd,
                                        bigA, 3 * Dd, 2 * Dd, attn_o, Dd,
                                        TQq, TQq, 1);
    k_gemm<<<dim3(ND / 128, Dd / 64), 256, 0, stream>>>(
        attn_o, d_sa_out[l], F32(base + 3), gout, ND, Dd, Dd, 0);
    k_add_ln<<<ND / 8, 256, 0, stream>>>(xd, gout, F32(base + 12),
                                         F32(base + 13), xd);
    // cross-attention: q from xd (wq = ca_in rows 0..D-1),
    // k,v from encoder memory (wk,wv = rows D..3D-1)
    k_gemm<<<dim3(ND / 128, Dd / 64), 256, 0, stream>>>(
        xd, d_ca_in[l], F32(base + 5), qd, ND, Dd, Dd, 0);
    k_gemm<<<dim3(NE / 128, (2 * Dd) / 64), 256, 0, stream>>>(
        xe, d_ca_in[l] + (size_t)Dd * Dd, F32(base + 5) + Dd, bigA, NE,
        2 * Dd, Dd, 0);
    k_attn<<<Bz * Hh, 256, 0, stream>>>(qd, Dd, 0, bigA, 2 * Dd, 0, bigA,
                                        2 * Dd, Dd, attn_o, Dd, TQq, Ss, 0);
    k_gemm<<<dim3(ND / 128, Dd / 64), 256, 0, stream>>>(
        attn_o, d_ca_out[l], F32(base + 7), gout, ND, Dd, Dd, 0);
    k_add_ln<<<ND / 8, 256, 0, stream>>>(xd, gout, F32(base + 14),
                                         F32(base + 15), xd);
    // FFN
    k_gemm<<<dim3(ND / 128, FFf / 64), 256, 0, stream>>>(
        xd, d_l1[l], F32(base + 9), bigA, ND, FFf, Dd, 1);
    k_gemm<<<dim3(ND / 128, Dd / 64), 256, 0, stream>>>(
        bigA, d_l2[l], F32(base + 11), gout, ND, Dd, FFf, 0);
    k_add_ln<<<ND / 8, 256, 0, stream>>>(xd, gout, F32(base + 16),
                                         F32(base + 17), xd);
  }

  // output projection + hist/IDM/fusion
  k_outproj<<<ND / 8, 256, 0, stream>>>(xd, F32(8), F32(9), out_tok);
  k_tail<<<1, 256, 0, stream>>>(F32(0), F32(3), out_tok, F32(10), F32(11),
                                (float*)d_out);
}
